// STMGCN_49435073577328
// MI455X (gfx1250) — matrix operand order repaired and checked by reading, not compiled, not run on hardware
//
#include <hip/hip_runtime.h>
#include <hip/hip_bf16.h>

// Problem constants (from reference)
#define NN      10000
#define NFEAT   512
#define NHID    64
#define NCLASS  10
#define KPAD    10016   // NN rounded up to multiple of 32 (K-tile of bf16 WMMA)
#define KTAIL   9984    // last K-step start; main loop is fully in-range below this

typedef __attribute__((ext_vector_type(16))) __bf16         v16bf;
typedef __attribute__((ext_vector_type(8)))  float          v8f;
typedef __attribute__((ext_vector_type(4)))  float          v4f;
typedef __attribute__((ext_vector_type(8)))  unsigned short v8us;

union BF16Frag {
    v16bf v;
    unsigned int w[8];     // packed pairs: w[i] = {elem 2i (lo16), elem 2i+1 (hi16)}
    v8us h[2];
};

__device__ __forceinline__ unsigned short f32_to_bf16_rne(float f) {
    unsigned int b = __float_as_uint(f);
    b += 0x7fffu + ((b >> 16) & 1u);   // round-to-nearest-even
    return (unsigned short)(b >> 16);
}

// Pack two f32 -> one dword of two bf16 (RNE). The high halves of the two
// rounded words are merged with a single v_perm_b32 (selector picks bytes
// {a.b2,a.b3,b.b2,b.b3}), avoiding 16-bit sub-register moves.
__device__ __forceinline__ unsigned int pack_bf16x2_rne(float lo, float hi) {
    unsigned int a = __float_as_uint(lo);
    unsigned int b = __float_as_uint(hi);
    a += 0x7fffu + ((a >> 16) & 1u);
    b += 0x7fffu + ((b >> 16) & 1u);
    // perm(s0, s1, sel): sel byte values 0-3 pick bytes of s1, 4-7 pick s0.
    // 0x07060302 -> D = {s1.b2, s1.b3, s0.b2, s0.b3} = {a_hi16, b_hi16}
    return __builtin_amdgcn_perm(b, a, 0x07060302u);
}

// ---------------------------------------------------------------------------
// Kernel 1: xw[b] = x @ W[b], stored TRANSPOSED as bf16: xwT[b][h][j], j padded
// to KPAD with zeros so the big GEMM's B-fragment loads need no K-tail guards.
// 1.3 GFLOP / 20 MB -> negligible; plain f32 FMA is fine here.
// ---------------------------------------------------------------------------
__global__ __launch_bounds__(128)
void xw_kernel(const float* __restrict__ x,
               const float* __restrict__ W1,
               const float* __restrict__ W2,
               unsigned short* __restrict__ xwT) {
    const int j = blockIdx.x;          // node index 0..KPAD-1
    const int h = threadIdx.x & 63;    // hidden unit
    const int b = threadIdx.x >> 6;    // branch
    const float* W = b ? W2 : W1;
    float acc = 0.0f;
    if (j < NN) {
        const float* xr = x + (size_t)j * NFEAT;
        #pragma unroll 8
        for (int f = 0; f < NFEAT; ++f)
            acc = fmaf(xr[f], W[f * NHID + h], acc);
    }
    xwT[(size_t)b * NHID * KPAD + (size_t)h * KPAD + j] =
        (j < NN) ? f32_to_bf16_rne(acc) : (unsigned short)0;
}

// ---------------------------------------------------------------------------
// Kernel 2: emb[b] = adj[b] @ xw[b] + bias[b]   (the 800 MB streaming GEMM)
// 8 waves/block; each wave: 16 rows x 64 cols via 4x v_wmma_f32_16x16x32_bf16
// per 32-wide K step. adj converted f32->bf16 in registers (no extra HBM
// traffic); f32 accumulation in the WMMA accumulators. Main loop is fully
// branchless; the single ragged K-step (9984..10015) is peeled and itself
// branchless (chunk0 uniformly in range, chunk1 uniformly zero).
// ---------------------------------------------------------------------------

// Load 16 adj f32 values (two 16B-aligned 8-float chunks) and pack the 16x32
// bf16 A fragment for this lane.
__device__ __forceinline__ void load_a_frag(BF16Frag& a, const float* __restrict__ aptr,
                                            int k0, int k1) {
    v4f f0 = *(const v4f*)(aptr + k0);
    v4f f1 = *(const v4f*)(aptr + k0 + 4);
    v4f f2 = *(const v4f*)(aptr + k1);
    v4f f3 = *(const v4f*)(aptr + k1 + 4);
    a.w[0] = pack_bf16x2_rne(f0[0], f0[1]);
    a.w[1] = pack_bf16x2_rne(f0[2], f0[3]);
    a.w[2] = pack_bf16x2_rne(f1[0], f1[1]);
    a.w[3] = pack_bf16x2_rne(f1[2], f1[3]);
    a.w[4] = pack_bf16x2_rne(f2[0], f2[1]);
    a.w[5] = pack_bf16x2_rne(f2[2], f2[3]);
    a.w[6] = pack_bf16x2_rne(f3[0], f3[1]);
    a.w[7] = pack_bf16x2_rne(f3[2], f3[3]);
}

__global__ __launch_bounds__(256)
void gcn_spmm_kernel(const float* __restrict__ adj1,
                     const float* __restrict__ adj2,
                     const unsigned short* __restrict__ xwT,
                     const float* __restrict__ b1,
                     const float* __restrict__ b2,
                     float* __restrict__ emb1,
                     float* __restrict__ emb2) {
    const int branch = blockIdx.y;
    const float* __restrict__ adj = branch ? adj2 : adj1;
    const unsigned short* __restrict__ xw = xwT + (size_t)branch * NHID * KPAD;
    const float* __restrict__ bias = branch ? b2 : b1;
    float* __restrict__ emb = branch ? emb2 : emb1;

    const int lane = threadIdx.x & 31;
    const int wave = threadIdx.x >> 5;
    const int m    = lane & 15;   // A-row / B-col / D-col within the 16x16 tile
    const int hl   = lane >> 4;   // lane half selects K sub-chunk / D row half
    const int rbase = blockIdx.x * 128 + wave * 16;

    int arow = rbase + m;
    if (arow >= NN) arow = NN - 1;            // clamped dup rows; stores guarded
    const float* __restrict__ aptr = adj + (size_t)arow * NN;

    v8f c[4] = {v8f{}, v8f{}, v8f{}, v8f{}};

    // -------- main loop: fully in-range, branchless --------
    #pragma unroll 2
    for (int jb = 0; jb < KTAIL; jb += 32) {
        const int k0 = jb + hl * 8;
        const int k1 = k0 + 16;

        // prefetch the A stream ~1KB ahead (global_prefetch_b8; speculative,
        // no LOADcnt, OOB translations silently dropped)
        __builtin_prefetch(aptr + k0 + 256, 0, 3);

        BF16Frag a;
        load_a_frag(a, aptr, k0, k1);

        const size_t bo = (size_t)m * KPAD + (size_t)(jb + hl * 8);
        BF16Frag bf[4];
        #pragma unroll
        for (int nb = 0; nb < 4; ++nb) {
            const unsigned short* p = xw + (size_t)nb * 16 * KPAD + bo;
            bf[nb].h[0] = *(const v8us*)(p);
            bf[nb].h[1] = *(const v8us*)(p + 16);
        }

        #pragma unroll
        for (int nb = 0; nb < 4; ++nb) {
            c[nb] = __builtin_amdgcn_wmma_f32_16x16x32_bf16(
                false, a.v, false, bf[nb].v, (short)0, c[nb], false, false);
        }
    }

    // -------- peeled ragged K-step (jb = 9984): K 9984..9999 real, rest 0.
    // chunk0 = [9984+hl*8, +8) is always in range; chunk1 >= 10000 always out.
    {
        const int k0 = KTAIL + hl * 8;
        BF16Frag a;
        v4f f0 = *(const v4f*)(aptr + k0);
        v4f f1 = *(const v4f*)(aptr + k0 + 4);
        a.w[0] = pack_bf16x2_rne(f0[0], f0[1]);
        a.w[1] = pack_bf16x2_rne(f0[2], f0[3]);
        a.w[2] = pack_bf16x2_rne(f1[0], f1[1]);
        a.w[3] = pack_bf16x2_rne(f1[2], f1[3]);
        a.w[4] = 0; a.w[5] = 0; a.w[6] = 0; a.w[7] = 0;

        const size_t bo = (size_t)m * KPAD + (size_t)(KTAIL + hl * 8);
        BF16Frag bf[4];
        #pragma unroll
        for (int nb = 0; nb < 4; ++nb) {
            const unsigned short* p = xw + (size_t)nb * 16 * KPAD + bo;
            bf[nb].h[0] = *(const v8us*)(p);
            bf[nb].h[1] = *(const v8us*)(p + 16);
        }
        #pragma unroll
        for (int nb = 0; nb < 4; ++nb) {
            c[nb] = __builtin_amdgcn_wmma_f32_16x16x32_bf16(
                false, a.v, false, bf[nb].v, (short)0, c[nb], false, false);
        }
    }

    // Epilogue: bias add + guarded store. D layout: element r of c[nb] is
    // row rbase + hl*8 + r, col nb*16 + m.
    #pragma unroll
    for (int nb = 0; nb < 4; ++nb) {
        const int h = nb * 16 + m;
        const float bv = bias[h];
        #pragma unroll
        for (int r = 0; r < 8; ++r) {
            const int row = rbase + hl * 8 + r;
            if (row < NN)
                emb[(size_t)row * NHID + h] = c[nb][r] + bv;
        }
    }
}

// ---------------------------------------------------------------------------
// Kernel 3: 2-view attention softmax fusion + DEC cluster assignment.
// q ~ (1 + d2/alpha)^(-0.72), normalized ((/2) cancels in normalization).
// ---------------------------------------------------------------------------
__global__ __launch_bounds__(256)
void fuse_dec_kernel(const float* __restrict__ emb1,
                     const float* __restrict__ emb2,
                     const float* __restrict__ attn_w,
                     const float* __restrict__ cluster,
                     float* __restrict__ out_emb,
                     float* __restrict__ out_q) {
    const int i = blockIdx.x * blockDim.x + threadIdx.x;
    if (i >= NN) return;
    const float* e1 = emb1 + (size_t)i * NHID;
    const float* e2 = emb2 + (size_t)i * NHID;

    float w1 = 0.0f, w2 = 0.0f;
    #pragma unroll 8
    for (int h = 0; h < NHID; ++h) {
        const float aw = attn_w[h];
        w1 = fmaf(e1[h], aw, w1);
        w2 = fmaf(e2[h], aw, w2);
    }
    const float mx = fmaxf(w1, w2);
    const float x1 = __expf(w1 - mx);
    const float x2 = __expf(w2 - mx);
    const float inv = 1.0f / (x1 + x2);
    const float be1 = x1 * inv, be2 = x2 * inv;

    float e[NHID];
    #pragma unroll 8
    for (int h = 0; h < NHID; ++h) {
        e[h] = be1 * e1[h] + be2 * e2[h];
        out_emb[(size_t)i * NHID + h] = e[h];
    }

    float q[NCLASS];
    float qs = 0.0f;
    #pragma unroll
    for (int k = 0; k < NCLASS; ++k) {
        const float* ck = cluster + k * NHID;
        float d2 = 0.0f;
        #pragma unroll 8
        for (int h = 0; h < NHID; ++h) {
            const float d = e[h] - ck[h];
            d2 = fmaf(d, d, d2);
        }
        // ((1+d2/a)^-1)^((a+1)/2) then ^(a+1): exponent -0.72 for a=0.2
        const float p = __powf(1.0f + d2 * 5.0f, -0.72f);
        q[k] = p;
        qs += p;
    }
    const float qinv = 1.0f / qs;
    #pragma unroll
    for (int k = 0; k < NCLASS; ++k)
        out_q[(size_t)i * NCLASS + k] = q[k] * qinv;
}

// ---------------------------------------------------------------------------
extern "C" void kernel_launch(void* const* d_in, const int* in_sizes, int n_in,
                              void* d_out, int out_size, void* d_ws, size_t ws_size,
                              hipStream_t stream) {
    const float* x       = (const float*)d_in[0];
    const float* adj1    = (const float*)d_in[1];
    const float* adj2    = (const float*)d_in[2];
    const float* W1      = (const float*)d_in[3];
    const float* b1      = (const float*)d_in[4];
    const float* W2      = (const float*)d_in[5];
    const float* b2      = (const float*)d_in[6];
    const float* attn_w  = (const float*)d_in[7];
    const float* cluster = (const float*)d_in[8];

    // Workspace layout: [xwT bf16: 2*64*KPAD] [emb1 f32: NN*64] [emb2 f32: NN*64]
    unsigned short* xwT = (unsigned short*)d_ws;
    const size_t XWT_ELEMS = (size_t)2 * NHID * KPAD;     // 2,564,096 B (16B-aligned)
    float* emb1 = (float*)(xwT + XWT_ELEMS);
    float* emb2 = emb1 + (size_t)NN * NHID;

    float* out_emb = (float*)d_out;                       // [NN, NHID]
    float* out_q   = out_emb + (size_t)NN * NHID;         // [NN, NCLASS]

    // 1) xw = x @ W (both branches), transposed bf16, zero-padded K
    xw_kernel<<<KPAD, 128, 0, stream>>>(x, W1, W2, xwT);

    // 2) emb[b] = adj[b] @ xw[b] + bias[b]  (WMMA bf16, f32 accumulate)
    dim3 g2((NN + 127) / 128, 2);
    gcn_spmm_kernel<<<g2, 256, 0, stream>>>(adj1, adj2, xwT, b1, b2, emb1, emb2);

    // 3) attention fusion + DEC soft assignment
    fuse_dec_kernel<<<(NN + 255) / 256, 256, 0, stream>>>(
        emb1, emb2, attn_w, cluster, out_emb, out_q);
}
